// EdgePooling_70935679861210
// MI455X (gfx1250) — matrix operand order repaired and checked by reading, not compiled, not run on hardware
//
#include <hip/hip_runtime.h>
#include <hip/hip_bf16.h>

// ---------------- problem constants ----------------
#define CH      128
#define NNODES  4096
#define NBATCH  64
#define NROWS   (NBATCH * NNODES)       // 262144
#define NEDGE   (NBATCH * (NNODES - 1)) // 262080
#define EPSLN   1e-5f

// ---------------- LDS tile geometry ----------------
#define TILE_ROWS 128
#define LDA 136   // f16 stride (multiple of 8 halves -> 16B aligned b128 frags, 68-dword rows)
#define LDH 132   // f32 stride of accumulator tile

#define SM_AH   0
#define SM_WH   (TILE_ROWS * LDA * 2)                // 34816
#define SM_HF   (SM_WH + TILE_ROWS * LDA * 2)        // 69632
#define SM_RED  (SM_HF + TILE_ROWS * LDH * 4)        // 137216
#define SM_PAR  (SM_RED + 512 * 4)                   // 139264
#define SM_TOTAL (SM_PAR + 6 * 128 * 4)              // 142336 bytes (2 blocks / 320KB WGP)

typedef __attribute__((ext_vector_type(16))) _Float16     v16h;
typedef __attribute__((ext_vector_type(8)))  float        v8f;
typedef __attribute__((ext_vector_type(4)))  unsigned int u32x4;
typedef __attribute__((ext_vector_type(2)))  unsigned int u32x2;
typedef __attribute__((ext_vector_type(4)))  float        f32x4;

union AFrag { v16h h; u32x4 q[2]; };
union H4    { u32x2 u; _Float16 h[4]; };
union U4H   { u32x4 q; _Float16 h[8]; };

// Async (ASYNCcnt-tracked) copy of a 128x128 f16 weight matrix from global into
// the padded LDS W tile: 8x b128 per thread, LDS written directly by the async
// path (no VGPR round trip). Dynamic-LDS base is offset 0 (no static LDS here),
// so the LDS destination address is just SM_WH plus the per-thread offset.
__device__ __forceinline__ void loadW_async(const _Float16* __restrict__ src, int tid) {
  const int r = tid >> 1, h = tid & 1;
  unsigned long long g = (unsigned long long)(size_t)(src + r * CH + h * 64);
  unsigned l = (unsigned)SM_WH + (unsigned)(r * (LDA * 2) + h * 128);
#pragma unroll
  for (int q = 0; q < 8; ++q) {
    asm volatile("global_load_async_to_lds_b128 %0, %1, off"
                 :: "v"(l + 16u * q), "v"(g + 16ull * q)
                 : "memory");
  }
}

__device__ __forceinline__ void wait_async0() {
  asm volatile("s_wait_asynccnt 0" ::: "memory");
}

// 128x128x128 GEMM: D = A(f16, LDS) x W(f16, LDS), f32 accum, +bias, -> Hf (LDS f32).
// 8 waves/block; wave (w&3) owns rows 32*(w&3).., wave (w>>2) owns cols 64*(w>>2)..
__device__ __forceinline__ void gemm128(const _Float16* __restrict__ Ah,
                                        const _Float16* __restrict__ Wh,
                                        float* __restrict__ Hf,
                                        const float* __restrict__ bias, int tid) {
  const int w = tid >> 5, lane = tid & 31;
  const int m0 = (w & 3) * 32, n0 = (w >> 2) * 64;
  const int ml = lane & 15, half = lane >> 4;

  const v8f vz = {0.f, 0.f, 0.f, 0.f, 0.f, 0.f, 0.f, 0.f};
  v8f acc[2][4];
#pragma unroll
  for (int mt = 0; mt < 2; ++mt)
#pragma unroll
    for (int nt = 0; nt < 4; ++nt) acc[mt][nt] = vz;

#pragma unroll
  for (int kt = 0; kt < 4; ++kt) {
    // A fragments (16x32 f16): lane = row; VGPR k-pattern 0-7 / 8-15 / 16-23 / 24-31
    AFrag a[2];
#pragma unroll
    for (int mt = 0; mt < 2; ++mt) {
      const _Float16* p = Ah + (m0 + mt * 16 + ml) * LDA + kt * 32 + half * 8;
      a[mt].q[0] = *(const u32x4*)p;        // K = base .. base+7   -> elems 0..7
      a[mt].q[1] = *(const u32x4*)(p + 16); // K = base+16..base+23 -> elems 8..15
    }
#pragma unroll
    for (int nt = 0; nt < 4; ++nt) {
      // B fragment (32x16 f16): lane%16 = N; elems 0-7 K = 8*half+s, elems 8-15 K = 16+8*half+s
      AFrag b;
      const _Float16* p = Wh + (kt * 32 + half * 8) * LDA + n0 + nt * 16 + ml;
#pragma unroll
      for (int s = 0; s < 8; ++s) {
        b.h[s] = p[s * LDA];
        b.h[8 + s] = p[(16 + s) * LDA];
      }
#pragma unroll
      for (int mt = 0; mt < 2; ++mt)
        acc[mt][nt] = __builtin_amdgcn_wmma_f32_16x16x32_f16(
            false, a[mt].h, false, b.h, (short)0, acc[mt][nt], false, false);
    }
  }

  // C/D layout: VGPR r -> M = r + 8*half, N = ml
#pragma unroll
  for (int mt = 0; mt < 2; ++mt)
#pragma unroll
    for (int nt = 0; nt < 4; ++nt) {
      const int col = n0 + nt * 16 + ml;
      const float bv = bias[col];
      const int rowb = m0 + mt * 16 + half * 8;
#pragma unroll
      for (int r = 0; r < 8; ++r)
        Hf[(rowb + r) * LDH + col] = acc[mt][nt][r] + bv;
    }
}

// LayerNorm + ELU over 128 cols, output f16 (LDS or global). 2 threads per row.
__device__ __forceinline__ void ln_elu(const float* __restrict__ Hf,
                                       const float* __restrict__ g,
                                       const float* __restrict__ be,
                                       _Float16* __restrict__ out, int ldo,
                                       float* __restrict__ red, int tid) {
  const int r = tid >> 1, h = tid & 1;
  const float* src = Hf + r * LDH + h * 64;
  float s = 0.f, q = 0.f;
#pragma unroll 8
  for (int c = 0; c < 64; ++c) { float v = src[c]; s += v; q += v * v; }
  red[tid] = s; red[256 + tid] = q;
  __syncthreads();
  const float ts = red[tid] + red[tid ^ 1];
  const float tq = red[256 + tid] + red[256 + (tid ^ 1)];
  const float mu = ts * (1.f / 128.f);
  const float var = tq * (1.f / 128.f) - mu * mu;
  const float rs = rsqrtf(var + EPSLN);
  _Float16* dst = out + (long)r * ldo + h * 64;
#pragma unroll 8
  for (int c = 0; c < 64; ++c) {
    const int col = h * 64 + c;
    float v = (src[c] - mu) * rs * g[col] + be[col];
    v = v > 0.f ? v : (__expf(v) - 1.f);
    dst[c] = (_Float16)v;
  }
  __syncthreads();
}

// ---------------- kernel 0: f32 -> f16 weights ----------------
__global__ void prep_kernel(const float* __restrict__ W1, const float* __restrict__ W2,
                            const float* __restrict__ Wr1, _Float16* __restrict__ w1h,
                            _Float16* __restrict__ w2h, _Float16* __restrict__ wr1h) {
  const int i = blockIdx.x * 256 + threadIdx.x;  // 16384 threads
  w1h[i] = (_Float16)W1[i];
  w2h[i] = (_Float16)W2[i];
  wr1h[i] = (_Float16)Wr1[i];
}

// ---------------- kernel 1: h2 = elu(ln(elu(ln(x@W1+b1))@W2+b2)) ----------------
__global__ void __launch_bounds__(256)
mlp2_kernel(const float* __restrict__ x, const _Float16* __restrict__ w1h,
            const _Float16* __restrict__ w2h, const float* __restrict__ b1,
            const float* __restrict__ g1, const float* __restrict__ be1,
            const float* __restrict__ b2, const float* __restrict__ g2,
            const float* __restrict__ be2, _Float16* __restrict__ h2) {
  extern __shared__ char smem[];
  _Float16* Ah = (_Float16*)(smem + SM_AH);
  _Float16* Wh = (_Float16*)(smem + SM_WH);
  float* Hf = (float*)(smem + SM_HF);
  float* red = (float*)(smem + SM_RED);
  float* par = (float*)(smem + SM_PAR);
  const int tid = threadIdx.x;
  const long row0 = (long)blockIdx.x * TILE_ROWS;

  loadW_async(w1h, tid);  // W1 -> LDS via async path, overlapped with x staging

  if (tid < 128) {
    par[tid] = b1[tid];       par[128 + tid] = g1[tid];  par[256 + tid] = be1[tid];
    par[384 + tid] = b2[tid]; par[512 + tid] = g2[tid];  par[640 + tid] = be2[tid];
  }
  {  // stage x tile -> f16 LDS
    const int r = tid >> 1, h = tid & 1;
    const f32x4* src = (const f32x4*)(x + (row0 + r) * CH + h * 64);
    _Float16* dst = Ah + r * LDA + h * 64;
#pragma unroll
    for (int q = 0; q < 16; ++q) {
      f32x4 v = src[q];
      H4 p;
      p.h[0] = (_Float16)v.x; p.h[1] = (_Float16)v.y;
      p.h[2] = (_Float16)v.z; p.h[3] = (_Float16)v.w;
      *(u32x2*)(dst + 4 * q) = p.u;
    }
  }
  wait_async0();
  __syncthreads();

  gemm128(Ah, Wh, Hf, par + 0, tid);
  __syncthreads();

  loadW_async(w2h, tid);  // W2 fetch overlaps the LayerNorm epilogue
  ln_elu(Hf, par + 128, par + 256, Ah, LDA, red, tid);  // layer1 act -> Ah (f16)
  wait_async0();
  __syncthreads();

  gemm128(Ah, Wh, Hf, par + 384, tid);
  __syncthreads();
  ln_elu(Hf, par + 512, par + 640, h2 + row0 * CH, CH, red, tid);  // -> global h2
}

// ---------------- kernel 2: edge max-pool + readout MLP ----------------
__global__ void __launch_bounds__(256)
edge_kernel(const _Float16* __restrict__ h2, const int* __restrict__ pidx,
            const _Float16* __restrict__ wr1h, const float* __restrict__ br1,
            const float* __restrict__ gr1, const float* __restrict__ ber1,
            const float* __restrict__ wr2, const float* __restrict__ br2,
            float* __restrict__ out) {
  extern __shared__ char smem[];
  _Float16* Ah = (_Float16*)(smem + SM_AH);
  _Float16* Wh = (_Float16*)(smem + SM_WH);
  float* Hf = (float*)(smem + SM_HF);
  float* red = (float*)(smem + SM_RED);
  float* par = (float*)(smem + SM_PAR);
  const int tid = threadIdx.x;
  const long e0 = (long)blockIdx.x * TILE_ROWS;

  loadW_async(wr1h, tid);  // Wr1 fetch overlaps the gather/max staging

  if (tid < 128) {
    par[tid] = br1[tid]; par[128 + tid] = gr1[tid];
    par[256 + tid] = ber1[tid]; par[384 + tid] = wr2[tid];
  }
  {  // gather child/parent rows, e = max -> f16 LDS tile
    const int r = tid >> 1, h = tid & 1;
    const long edge = e0 + r;
    _Float16* dst = Ah + r * LDA + h * 64;
    if (edge < NEDGE) {
      const int bb = (int)(edge / (NNODES - 1));
      const int j = (int)(edge - (long)bb * (NNODES - 1));
      const u32x4* cs = (const u32x4*)(h2 + ((long)bb * NNODES + j) * CH + h * 64);
      const u32x4* ps = (const u32x4*)(h2 + (long)pidx[edge] * CH + h * 64);
#pragma unroll
      for (int q = 0; q < 8; ++q) {
        U4H c, p, m;
        c.q = cs[q]; p.q = ps[q];
#pragma unroll
        for (int e2 = 0; e2 < 8; ++e2) m.h[e2] = (c.h[e2] > p.h[e2]) ? c.h[e2] : p.h[e2];
        *(u32x4*)(dst + 8 * q) = m.q;
      }
    } else {
      const u32x4 z = {0u, 0u, 0u, 0u};
#pragma unroll
      for (int q = 0; q < 8; ++q) *(u32x4*)(dst + 8 * q) = z;
    }
  }
  wait_async0();
  __syncthreads();

  gemm128(Ah, Wh, Hf, par + 0, tid);
  __syncthreads();

  {  // LN + ELU + dot(Wr2) + br2 -> out[edge]
    const int r = tid >> 1, h = tid & 1;
    const float* src = Hf + r * LDH + h * 64;
    float s = 0.f, q = 0.f;
#pragma unroll 8
    for (int c = 0; c < 64; ++c) { float v = src[c]; s += v; q += v * v; }
    red[tid] = s; red[256 + tid] = q;
    __syncthreads();
    const float ts = red[tid] + red[tid ^ 1];
    const float tq = red[256 + tid] + red[256 + (tid ^ 1)];
    const float mu = ts * (1.f / 128.f);
    const float var = tq * (1.f / 128.f) - mu * mu;
    const float rs = rsqrtf(var + EPSLN);
    float dot = 0.f;
#pragma unroll 8
    for (int c = 0; c < 64; ++c) {
      const int col = h * 64 + c;
      float v = (src[c] - mu) * rs * par[128 + col] + par[256 + col];
      v = v > 0.f ? v : (__expf(v) - 1.f);
      dot += v * par[384 + col];
    }
    __syncthreads();
    red[tid] = dot;
    __syncthreads();
    if (h == 0) {
      const long edge = e0 + r;
      if (edge < NEDGE) out[edge] = red[tid] + red[tid ^ 1] + br2[0];
    }
  }
}

// ---------------- kernel 3: out -= logsumexp(out, axis=1) ----------------
__global__ void lse_kernel(float* __restrict__ out) {
  __shared__ float red[256];
  const int t = threadIdx.x;
  float* row = out + (long)blockIdx.x * (NNODES - 1);
  float m = -INFINITY;
  for (int i = t; i < NNODES - 1; i += 256) m = fmaxf(m, row[i]);
  red[t] = m;
  __syncthreads();
  for (int s = 128; s > 0; s >>= 1) {
    if (t < s) red[t] = fmaxf(red[t], red[t + s]);
    __syncthreads();
  }
  m = red[0];
  __syncthreads();
  float sum = 0.f;
  for (int i = t; i < NNODES - 1; i += 256) sum += __expf(row[i] - m);
  red[t] = sum;
  __syncthreads();
  for (int s = 128; s > 0; s >>= 1) {
    if (t < s) red[t] += red[t + s];
    __syncthreads();
  }
  const float lse = m + __logf(red[0]);
  __syncthreads();
  for (int i = t; i < NNODES - 1; i += 256) row[i] -= lse;
}

// ---------------- host launch ----------------
extern "C" void kernel_launch(void* const* d_in, const int* in_sizes, int n_in,
                              void* d_out, int out_size, void* d_ws, size_t ws_size,
                              hipStream_t stream) {
  const float* x   = (const float*)d_in[0];
  const int* pidx  = (const int*)d_in[1];
  const float* W1  = (const float*)d_in[2];
  const float* b1  = (const float*)d_in[3];
  const float* g1  = (const float*)d_in[4];
  const float* be1 = (const float*)d_in[5];
  const float* W2  = (const float*)d_in[6];
  const float* b2  = (const float*)d_in[7];
  const float* g2  = (const float*)d_in[8];
  const float* be2 = (const float*)d_in[9];
  const float* Wr1 = (const float*)d_in[10];
  const float* br1 = (const float*)d_in[11];
  const float* gr1 = (const float*)d_in[12];
  const float* ber1= (const float*)d_in[13];
  const float* Wr2 = (const float*)d_in[14];
  const float* br2 = (const float*)d_in[15];
  float* out = (float*)d_out;

  // workspace layout: [W1h | W2h | Wr1h | h2]
  _Float16* w1h  = (_Float16*)d_ws;
  _Float16* w2h  = w1h + 128 * 128;
  _Float16* wr1h = w2h + 128 * 128;
  _Float16* h2   = wr1h + 128 * 128;  // NROWS*128 f16 = 64 MB

  (void)in_sizes; (void)n_in; (void)out_size; (void)ws_size;

  hipFuncSetAttribute((const void*)mlp2_kernel,
                      hipFuncAttributeMaxDynamicSharedMemorySize, SM_TOTAL);
  hipFuncSetAttribute((const void*)edge_kernel,
                      hipFuncAttributeMaxDynamicSharedMemorySize, SM_TOTAL);

  prep_kernel<<<64, 256, 0, stream>>>(W1, W2, Wr1, w1h, w2h, wr1h);
  mlp2_kernel<<<NROWS / TILE_ROWS, 256, SM_TOTAL, stream>>>(
      x, w1h, w2h, b1, g1, be1, b2, g2, be2, h2);
  edge_kernel<<<(NEDGE + TILE_ROWS - 1) / TILE_ROWS, 256, SM_TOTAL, stream>>>(
      h2, pidx, wr1h, br1, gr1, ber1, Wr2, br2, out);
  lse_kernel<<<NBATCH, 256, 0, stream>>>(out);
}
